// GeomolGNNOGBFeatRandomNonShared_76038101008805
// MI455X (gfx1250) — matrix operand order repaired and checked by reading, not compiled, not run on hardware
//
#include <hip/hip_runtime.h>
#include <hip/hip_bf16.h>

// ---------------------------------------------------------------------------
// GeoMol GNN on MI455X (gfx1250): all 11 MLPs run as bf16 WMMA GEMMs with f32
// accumulation (v_wmma_f32_16x16x32_bf16). Workload is HBM-bound (~10 GB
// streamed vs 930 GFLOP), so the goal is: WMMA for the matmuls so compute
// never becomes the bottleneck, contiguous 16B/32B per-lane fragment loads,
// fused bias / BN+ReLU / bf16 conversion epilogues.
// ---------------------------------------------------------------------------

typedef __attribute__((ext_vector_type(16))) __bf16 v16bf;
typedef __attribute__((ext_vector_type(8)))  float  v8f;

#define HID   300
#define NPAD  320     // HID padded to multiple of 32 for WMMA N/K
#define RVD   10
#define BN_EPS 1e-5f

static __device__ __forceinline__ __bf16 to_bf16(float f) { return (__bf16)f; }

// ---------------------------------------------------------------------------
// GEMM: C[M,HID] = A_bf16[M,Kpad] x Wt_bf16[NPAD,Kpad]^T + bias
// Wt is stored transposed (row = output column n, contiguous in K) so each
// lane's B fragment (N = n0+lane&15) is two 16-byte chunks matching A:
// K = k0+8g..+7 and K = k0+16+8g..+7. A fragment per lane (row M = lane&15) is two
// 16-byte chunks: K = k0+8g..+7 and K = k0+16+8g..+7 (CDNA5 16-bit A layout).
// Block = 256 threads = 8 waves; wave w owns rows m0+16w..+15 and a 160-col
// strip (10 WMMA tiles, 80 f32 accumulator VGPRs). gridDim.y = 2 strips.
// ---------------------------------------------------------------------------
__global__ __launch_bounds__(256)
void gemm_bf16_wmma(const __bf16* __restrict__ A, const __bf16* __restrict__ Wt,
                    const float* __restrict__ bias, float* __restrict__ C,
                    int M, int Kpad)
{
    const int lane = threadIdx.x & 31;
    const int wave = threadIdx.x >> 5;
    const int g    = lane >> 4;      // half-wave group
    const int l15  = lane & 15;

    const int m0 = blockIdx.x * 128 + wave * 16;
    const int n0 = blockIdx.y * 160;

    int arow = m0 + l15;
    if (arow >= M) arow = M - 1;     // clamp: stores are guarded, loads stay in-bounds
    const __bf16* __restrict__ aptr = A + (size_t)arow * Kpad + 8 * g;

    v8f acc[10];
    const v8f vzero = {0.f,0.f,0.f,0.f,0.f,0.f,0.f,0.f};
#pragma unroll
    for (int t = 0; t < 10; ++t) acc[t] = vzero;

    for (int k0 = 0; k0 < Kpad; k0 += 32) {
        union { uint4 u[2]; v16bf v; } af;
        af.u[0] = *(const uint4*)(aptr + k0);
        af.u[1] = *(const uint4*)(aptr + k0 + 16);
#pragma unroll
        for (int t = 0; t < 10; ++t) {
            const __bf16* __restrict__ bptr =
                Wt + (size_t)(n0 + t * 16 + l15) * Kpad + k0 + 8 * g;
            union { uint4 u[2]; v16bf v; } bf;
            bf.u[0] = *(const uint4*)(bptr);
            bf.u[1] = *(const uint4*)(bptr + 16);
            acc[t] = __builtin_amdgcn_wmma_f32_16x16x32_bf16(
                false, af.v, false, bf.v, (short)0, acc[t], false, false);
        }
    }

    // C/D layout: VGPR r -> row m0 + r + 8*g, col n0 + t*16 + (lane&15)
#pragma unroll
    for (int t = 0; t < 10; ++t) {
        int col = n0 + t * 16 + l15;
        if (col >= HID) continue;
        float b = bias[col];
#pragma unroll
        for (int r = 0; r < 8; ++r) {
            int rowi = m0 + r + 8 * g;
            if (rowi < M) C[(size_t)rowi * HID + col] = acc[t][r] + b;
        }
    }
}

// ------------------------- weight transpose + pad --------------------------
__global__ void prep_weight(const float* __restrict__ W, __bf16* __restrict__ Wt,
                            int K, int Kpad)
{
    int idx = blockIdx.x * blockDim.x + threadIdx.x;
    int total = NPAD * Kpad;
    if (idx >= total) return;
    int n = idx / Kpad, k = idx - n * Kpad;
    float v = (n < HID && k < K) ? W[(size_t)k * HID + n] : 0.f;
    Wt[idx] = to_bf16(v);
}

// ------------------------- batchnorm statistics ----------------------------
__global__ __launch_bounds__(320)
void col_stats(const float* __restrict__ X, int M, float* __restrict__ stats)
{
    const int c  = threadIdx.x;
    const int r0 = blockIdx.x * 512;
    if (c >= HID) return;
    int r1 = r0 + 512; if (r1 > M) r1 = M;
    float s = 0.f, s2 = 0.f;
    for (int r = r0; r < r1; ++r) {
        float v = X[(size_t)r * HID + c];
        s += v; s2 += v * v;
    }
    atomicAdd(&stats[c], s);
    atomicAdd(&stats[NPAD + c], s2);
}

// --------------------- BN (train-mode) + ReLU + to bf16 --------------------
__global__ void bn_relu_bf16(const float* __restrict__ X, const float* __restrict__ stats,
                             const float* __restrict__ gamma, const float* __restrict__ beta,
                             __bf16* __restrict__ Y, int M)
{
    size_t idx = (size_t)blockIdx.x * blockDim.x + threadIdx.x;
    size_t total = (size_t)M * NPAD;
    if (idx >= total) return;
    int c = (int)(idx % NPAD);
    size_t m = idx / NPAD;
    float v = 0.f;
    if (c < HID) {
        float invM = 1.0f / (float)M;
        float mean = stats[c] * invM;
        float var  = stats[NPAD + c] * invM - mean * mean;
        float x = X[m * HID + c];
        float y = gamma[c] * (x - mean) * rsqrtf(var + BN_EPS) + beta[c];
        v = y > 0.f ? y : 0.f;
    }
    Y[idx] = to_bf16(v);
}

// ------------------------- feature encoders --------------------------------
__global__ void embed_nodes(const int* __restrict__ x, const float* __restrict__ rx,
                            const float* t0, const float* t1, const float* t2,
                            const float* t3, const float* t4, const float* t5,
                            const float* t6, const float* t7, const float* t8,
                            __bf16* __restrict__ out, int N)
{
    size_t idx = (size_t)blockIdx.x * blockDim.x + threadIdx.x;
    size_t total = (size_t)N * NPAD;
    if (idx >= total) return;
    int c = (int)(idx % NPAD);
    size_t n = idx / NPAD;
    float v = 0.f;
    if (c < HID) {
        const int* xi = x + n * 9;
        v  = t0[(size_t)xi[0] * HID + c] + t1[(size_t)xi[1] * HID + c]
           + t2[(size_t)xi[2] * HID + c] + t3[(size_t)xi[3] * HID + c]
           + t4[(size_t)xi[4] * HID + c] + t5[(size_t)xi[5] * HID + c]
           + t6[(size_t)xi[6] * HID + c] + t7[(size_t)xi[7] * HID + c]
           + t8[(size_t)xi[8] * HID + c];
    } else if (c < HID + RVD) {
        v = rx[n * RVD + (c - HID)];
    }
    out[idx] = to_bf16(v);
}

__global__ void embed_edges(const int* __restrict__ ea, const float* __restrict__ re,
                            const float* t0, const float* t1, const float* t2,
                            __bf16* __restrict__ out, int E)
{
    size_t idx = (size_t)blockIdx.x * blockDim.x + threadIdx.x;
    size_t total = (size_t)E * NPAD;
    if (idx >= total) return;
    int c = (int)(idx % NPAD);
    size_t e = idx / NPAD;
    float v = 0.f;
    if (c < HID) {
        const int* ei = ea + e * 3;
        v = t0[(size_t)ei[0] * HID + c] + t1[(size_t)ei[1] * HID + c]
          + t2[(size_t)ei[2] * HID + c];
    } else if (c < HID + RVD) {
        v = re[e * RVD + (c - HID)];
    }
    out[idx] = to_bf16(v);
}

// ------------------------- concat builders (bf16, K-padded) ----------------
#define W3 928   // 3*HID=900 -> 928
#define W2C 608  // 2*HID=600 -> 608

__global__ void concat_edge(const float* __restrict__ h, const float* __restrict__ e,
                            const int* __restrict__ row, const int* __restrict__ col,
                            __bf16* __restrict__ out, int E)
{
    size_t idx = (size_t)blockIdx.x * blockDim.x + threadIdx.x;
    size_t total = (size_t)E * W3;
    if (idx >= total) return;
    int c = (int)(idx % W3);
    size_t ei = idx / W3;
    float v = 0.f;
    if (c < HID)            v = h[(size_t)row[ei] * HID + c];
    else if (c < 2 * HID)   v = h[(size_t)col[ei] * HID + (c - HID)];
    else if (c < 3 * HID)   v = e[ei * HID + (c - 2 * HID)];
    out[idx] = to_bf16(v);
}

__global__ void concat_he(const float* __restrict__ h, const float* __restrict__ e,
                          const int* __restrict__ row, __bf16* __restrict__ out, int E)
{
    size_t idx = (size_t)blockIdx.x * blockDim.x + threadIdx.x;
    size_t total = (size_t)E * W2C;
    if (idx >= total) return;
    int c = (int)(idx % W2C);
    size_t ei = idx / W2C;
    float v = 0.f;
    if (c < HID)          v = h[(size_t)row[ei] * HID + c];
    else if (c < 2 * HID) v = e[ei * HID + (c - HID)];
    out[idx] = to_bf16(v);
}

__global__ void concat_hagg(const float* __restrict__ h, const float* __restrict__ agg,
                            __bf16* __restrict__ out, int N)
{
    size_t idx = (size_t)blockIdx.x * blockDim.x + threadIdx.x;
    size_t total = (size_t)N * W2C;
    if (idx >= total) return;
    int c = (int)(idx % W2C);
    size_t n = idx / W2C;
    float v = 0.f;
    if (c < HID)          v = h[n * HID + c];
    else if (c < 2 * HID) v = agg[n * HID + (c - HID)];
    out[idx] = to_bf16(v);
}

// ------------------------- segment mean ------------------------------------
__global__ void count_edges(const int* __restrict__ col, float* __restrict__ cnt, int E)
{
    int i = blockIdx.x * blockDim.x + threadIdx.x;
    if (i < E) atomicAdd(&cnt[col[i]], 1.0f);
}

__global__ void scatter_add(const float* __restrict__ m, const int* __restrict__ col,
                            float* __restrict__ agg, int E)
{
    size_t idx = (size_t)blockIdx.x * blockDim.x + threadIdx.x;
    size_t total = (size_t)E * HID;
    if (idx >= total) return;
    size_t e = idx / HID;
    int c = (int)(idx - e * HID);
    atomicAdd(&agg[(size_t)col[e] * HID + c], m[idx]);
}

__global__ void div_cnt(float* __restrict__ agg, const float* __restrict__ cnt, int N)
{
    size_t idx = (size_t)blockIdx.x * blockDim.x + threadIdx.x;
    size_t total = (size_t)N * HID;
    if (idx >= total) return;
    agg[idx] /= fmaxf(cnt[idx / HID], 1.0f);
}

// ---------------------------------------------------------------------------
extern "C" void kernel_launch(void* const* d_in, const int* in_sizes, int n_in,
                              void* d_out, int out_size, void* d_ws, size_t ws_size,
                              hipStream_t stream)
{
    const int N = in_sizes[0] / 9;
    const int E = in_sizes[1] / 2;

    const int*   x     = (const int*)d_in[0];
    const int*   eidx  = (const int*)d_in[1];
    const int*   rowp  = eidx;
    const int*   colp  = eidx + E;
    const int*   eattr = (const int*)d_in[2];
    const float* randx = (const float*)d_in[3];
    const float* rande = (const float*)d_in[4];
    const float* aemb[9]; for (int i = 0; i < 9; ++i) aemb[i] = (const float*)d_in[5 + i];
    const float* bemb[3]; for (int i = 0; i < 3; ++i) bemb[i] = (const float*)d_in[14 + i];

    // JAX pytree dict flattening is key-sorted:
    // params: atom_emb, bond_emb, edge_init, layers, node_init
    // each MLP dict: W1, W2, b1, b2, be1, g1
    struct MlpP { const float *W1, *W2, *b1, *b2, *be1, *g1; int K1; };
    auto mlp_from = [&](int base, int K1) {
        MlpP p;
        p.W1  = (const float*)d_in[base + 0];
        p.W2  = (const float*)d_in[base + 1];
        p.b1  = (const float*)d_in[base + 2];
        p.b2  = (const float*)d_in[base + 3];
        p.be1 = (const float*)d_in[base + 4];
        p.g1  = (const float*)d_in[base + 5];
        p.K1  = K1;
        return p;
    };
    MlpP edge_init = mlp_from(17, HID + RVD);
    MlpP Lp[3][3];
    for (int i = 0; i < 3; ++i) {
        Lp[i][0] = mlp_from(23 + 18 * i + 0,  3 * HID);  // edge
        Lp[i][1] = mlp_from(23 + 18 * i + 6,  2 * HID);  // node1
        Lp[i][2] = mlp_from(23 + 18 * i + 12, 2 * HID);  // node2
    }
    MlpP node_init = mlp_from(77, HID + RVD);

    // ---- carve workspace ----
    char* ws = (char*)d_ws;
    size_t off = 0;
    auto carve = [&](size_t bytes) -> char* {
        char* p = ws + off;
        off = (off + bytes + 255) & ~(size_t)255;
        return p;
    };
    float*  h     = (float*)carve((size_t)N * HID * 4);
    float*  e     = (float*)carve((size_t)E * HID * 4);
    float*  cnt   = (float*)carve((size_t)N * 4);
    float*  agg   = (float*)carve((size_t)N * HID * 4);
    __bf16* big   = (__bf16*)carve((size_t)E * W3 * 2);    // concat / embed inputs
    float*  mid   = (float*)carve((size_t)E * HID * 4);    // gemm1 out / m
    __bf16* midb  = (__bf16*)carve((size_t)E * NPAD * 2);  // post BN+ReLU
    float*  stats = (float*)carve(2 * NPAD * 4);

    struct MlpW { __bf16 *Wt1, *Wt2; int Kpad1; };
    auto prep = [&](const MlpP& p) {
        MlpW w;
        w.Kpad1 = ((p.K1 + 31) / 32) * 32;
        w.Wt1 = (__bf16*)carve((size_t)NPAD * w.Kpad1 * 2);
        w.Wt2 = (__bf16*)carve((size_t)NPAD * NPAD * 2);
        int tot1 = NPAD * w.Kpad1;
        prep_weight<<<(tot1 + 255) / 256, 256, 0, stream>>>(p.W1, w.Wt1, p.K1, w.Kpad1);
        int tot2 = NPAD * NPAD;
        prep_weight<<<(tot2 + 255) / 256, 256, 0, stream>>>(p.W2, w.Wt2, HID, NPAD);
        return w;
    };
    MlpW w_node = prep(node_init);
    MlpW w_edge = prep(edge_init);
    MlpW wl[3][3];
    for (int i = 0; i < 3; ++i)
        for (int j = 0; j < 3; ++j) wl[i][j] = prep(Lp[i][j]);

    auto run_mlp = [&](const __bf16* A, int M, const MlpP& p, const MlpW& w, float* dest) {
        dim3 gg((M + 127) / 128, 2);
        gemm_bf16_wmma<<<gg, 256, 0, stream>>>(A, w.Wt1, p.b1, mid, M, w.Kpad1);
        hipMemsetAsync(stats, 0, 2 * NPAD * 4, stream);
        col_stats<<<(M + 511) / 512, 320, 0, stream>>>(mid, M, stats);
        size_t tot = (size_t)M * NPAD;
        bn_relu_bf16<<<(unsigned)((tot + 255) / 256), 256, 0, stream>>>(mid, stats, p.g1, p.be1, midb, M);
        gemm_bf16_wmma<<<gg, 256, 0, stream>>>(midb, w.Wt2, p.b2, dest, M, NPAD);
    };

    // ---- degree counts ----
    hipMemsetAsync(cnt, 0, (size_t)N * 4, stream);
    count_edges<<<(E + 255) / 256, 256, 0, stream>>>(colp, cnt, E);

    // ---- encoders + init MLPs ----
    size_t totn = (size_t)N * NPAD;
    embed_nodes<<<(unsigned)((totn + 255) / 256), 256, 0, stream>>>(
        x, randx, aemb[0], aemb[1], aemb[2], aemb[3], aemb[4], aemb[5], aemb[6], aemb[7], aemb[8],
        big, N);
    run_mlp(big, N, node_init, w_node, h);

    size_t tote = (size_t)E * NPAD;
    embed_edges<<<(unsigned)((tote + 255) / 256), 256, 0, stream>>>(
        eattr, rande, bemb[0], bemb[1], bemb[2], big, E);
    run_mlp(big, E, edge_init, w_edge, e);

    // ---- message-passing layers ----
    for (int l = 0; l < 3; ++l) {
        size_t t1 = (size_t)E * W3;
        concat_edge<<<(unsigned)((t1 + 255) / 256), 256, 0, stream>>>(h, e, rowp, colp, big, E);
        run_mlp(big, E, Lp[l][0], wl[l][0], e);

        size_t t2 = (size_t)E * W2C;
        concat_he<<<(unsigned)((t2 + 255) / 256), 256, 0, stream>>>(h, e, rowp, big, E);
        run_mlp(big, E, Lp[l][1], wl[l][1], mid);     // m -> mid

        hipMemsetAsync(agg, 0, (size_t)N * HID * 4, stream);
        size_t t3 = (size_t)E * HID;
        scatter_add<<<(unsigned)((t3 + 255) / 256), 256, 0, stream>>>(mid, colp, agg, E);
        size_t t4 = (size_t)N * HID;
        div_cnt<<<(unsigned)((t4 + 255) / 256), 256, 0, stream>>>(agg, cnt, N);

        size_t t5 = (size_t)N * W2C;
        concat_hagg<<<(unsigned)((t5 + 255) / 256), 256, 0, stream>>>(h, agg, big, N);
        run_mlp(big, N, Lp[l][2], wl[l][2], h);
    }

    // ---- outputs: (h, e) concatenated flat ----
    hipMemcpyAsync(d_out, h, (size_t)N * HID * 4, hipMemcpyDeviceToDevice, stream);
    hipMemcpyAsync((float*)d_out + (size_t)N * HID, e, (size_t)E * HID * 4,
                   hipMemcpyDeviceToDevice, stream);
}
